// BatchedExpertPool_352187318778
// MI455X (gfx1250) — matrix operand order repaired and checked by reading, not compiled, not run on hardware
//
#include <hip/hip_runtime.h>
#include <math.h>

// B=4, L=2048, D=512, E=8, H=1024. M = B*L = 8192.
#define M_TOT 8192
#define D_DIM 512
#define E_NUM 8
#define H_DIM 1024

typedef __attribute__((ext_vector_type(16))) _Float16 v16h;
typedef __attribute__((ext_vector_type(8)))  float    v8f;

// ---------------------------------------------------------------------------
// Pack kernels: f32 -> f16 directly in WMMA fragment layout.
// A-fragment (16x32 f16): lanes 0-15 -> M=lane, elems {K0..7, K16..23};
//                         lanes 16-31 -> M=lane-16, elems {K8..15, K24..31}.
// B-fragment (32x16 f16): lanes 0-15 -> N=lane, elems {K0..7, K16..23};
//                         lanes 16-31 -> N=lane-16, elems {K8..15, K24..31}.
// One tile = 32 lanes x 16 halves = 1 KB; per-lane load is 32B contiguous.
// ---------------------------------------------------------------------------

// xp tiles ordered [mt(512)][kt(16)]
__global__ __launch_bounds__(256) void pack_x_kernel(const float* __restrict__ x,
                                                     _Float16* __restrict__ xp) {
    int wave = (blockIdx.x * blockDim.x + threadIdx.x) >> 5;
    int lane = threadIdx.x & 31;
    int kt = wave & 15;
    int mt = wave >> 4;
    int m  = lane & 15;
    int koff = (lane < 16) ? 0 : 8;
    const float* src = x + (size_t)(mt * 16 + m) * D_DIM + kt * 32;
    v16h v;
#pragma unroll
    for (int i = 0; i < 8; ++i) {
        v[i]     = (_Float16)src[koff + i];
        v[i + 8] = (_Float16)src[16 + koff + i];
    }
    *((v16h*)xp + (size_t)wave * 32 + lane) = v;
}

// w1p tiles ordered [e(8)][kt(16)][nt(64)]  (K = D, N = H)
__global__ __launch_bounds__(256) void pack_w1_kernel(const float* __restrict__ w1,
                                                      _Float16* __restrict__ w1p) {
    int wave = (blockIdx.x * blockDim.x + threadIdx.x) >> 5;
    int lane = threadIdx.x & 31;
    int nt = wave & 63;
    int kt = (wave >> 6) & 15;
    int e  = wave >> 10;
    int n  = lane & 15;
    int kb = (lane < 16) ? 0 : 8;
    v16h v;
#pragma unroll
    for (int i = 0; i < 16; ++i) {
        int k = kb + (i & 7) + ((i >> 3) << 4);
        v[i] = (_Float16)w1[((size_t)e * D_DIM + kt * 32 + k) * H_DIM + nt * 16 + n];
    }
    *((v16h*)w1p + (size_t)wave * 32 + lane) = v;
}

// w2p tiles ordered [e(8)][kt(32)][nt(32)]  (K = H, N = D)
__global__ __launch_bounds__(256) void pack_w2_kernel(const float* __restrict__ w2,
                                                      _Float16* __restrict__ w2p) {
    int wave = (blockIdx.x * blockDim.x + threadIdx.x) >> 5;
    int lane = threadIdx.x & 31;
    int nt = wave & 31;
    int kt = (wave >> 5) & 31;
    int e  = wave >> 10;
    int n  = lane & 15;
    int kb = (lane < 16) ? 0 : 8;
    v16h v;
#pragma unroll
    for (int i = 0; i < 16; ++i) {
        int k = kb + (i & 7) + ((i >> 3) << 4);
        v[i] = (_Float16)w2[((size_t)e * H_DIM + kt * 32 + k) * D_DIM + nt * 16 + n];
    }
    *((v16h*)w2p + (size_t)wave * 32 + lane) = v;
}

// ---------------------------------------------------------------------------
// Fused kernel: per 32-row tile, loop experts x H-chunks:
//   h = x @ w1[e] (WMMA)  -> +b1 -> exact GELU -> * weights[row,e] -> f16 LDS
//   acc += h @ w2[e]      (WMMA, persistent f32 accumulators)
// Epilogue: acc + sum_e weights[row,e]*b2[e,:]  -> out (f32)
// ---------------------------------------------------------------------------
__global__ __launch_bounds__(256) void moe_fused_kernel(
    const _Float16* __restrict__ xp, const _Float16* __restrict__ w1p,
    const _Float16* __restrict__ w2p, const float* __restrict__ weights,
    const float* __restrict__ b1, const float* __restrict__ b2,
    float* __restrict__ out) {

    __shared__ __align__(32) _Float16 hbuf[2 * 2 * 32 * 16]; // [rt][kt][laneA][16] A-layout, 4KB
    __shared__ float wtile[32];                              // weights[row, e] for this expert

    const int tid  = threadIdx.x;
    const int lane = tid & 31;
    const int w    = tid >> 5;          // wave 0..7
    const int m0   = blockIdx.x * 32;   // first row of tile
    const int mt0  = blockIdx.x * 2;    // xp row-tile base
    const int rt1  = w >> 2;            // GEMM1: row-tile 0..1
    const int ct1  = w & 3;             // GEMM1: col-tile within 64-wide chunk
    const int rt2  = w >> 2;            // GEMM2: row-tile (persistent acc rows)
    const int ntb  = (w & 3) * 8;       // GEMM2: base N-tile (of 32)
    const int hi   = lane >> 4;         // 0/1 half-wave (C-layout M offset 0/8)
    const int nn   = lane & 15;

    v8f acc[8];
#pragma unroll
    for (int j = 0; j < 8; ++j) acc[j] = v8f{};

    const v16h* Abase = (const v16h*)xp + ((size_t)(mt0 + rt1) * 16) * 32 + lane;

    for (int e = 0; e < E_NUM; ++e) {
        if (tid < 32) wtile[tid] = weights[(size_t)(m0 + tid) * E_NUM + e];
        __syncthreads();

        for (int hc = 0; hc < 16; ++hc) {
            // ---- GEMM1: one 16x16 h-tile per wave, K = 512 ----
            v8f acc1 = v8f{};
            const v16h* Bbase = (const v16h*)w1p +
                ((size_t)(e * 16) * 64 + hc * 4 + ct1) * 32 + lane;
#pragma unroll
            for (int kt = 0; kt < 16; ++kt) {
                v16h a = Abase[kt * 32];
                v16h b = Bbase[(size_t)kt * 64 * 32];
                acc1 = __builtin_amdgcn_wmma_f32_16x16x32_f16(
                    false, a, false, b, (short)0, acc1, false, false);
            }

            __syncthreads(); // previous chunk's GEMM2 reads of hbuf complete

            // ---- bias + exact GELU + expert-weight scale -> hbuf (A-layout) ----
            const float b1v = b1[e * H_DIM + hc * 64 + ct1 * 16 + nn];
            const int c = ct1 * 16 + nn;   // column within 64-wide chunk
            const int kt2 = c >> 5;
            const int kk  = c & 31;
            const int sub = kk & 15;
            const int lh  = (sub >> 3) & 1;
            const int elem = (sub & 7) + 8 * (kk >> 4);
#pragma unroll
            for (int j = 0; j < 8; ++j) {
                const int mrow = j + hi * 8;
                float v = acc1[j] + b1v;
                float g = 0.5f * v * (1.0f + erff(v * 0.70710678118654752f));
                g *= wtile[rt1 * 16 + mrow];
                const int laneA = mrow + 16 * lh;
                hbuf[(((rt1 * 2 + kt2) * 32) + laneA) * 16 + elem] = (_Float16)g;
            }
            __syncthreads();

            // ---- GEMM2: acc += h_chunk (32xK=64) @ w2[e] chunk ----
            v16h A2[2];
#pragma unroll
            for (int k2 = 0; k2 < 2; ++k2)
                A2[k2] = *(const v16h*)&hbuf[((rt2 * 2 + k2) * 32 + lane) * 16];

#pragma unroll
            for (int j = 0; j < 8; ++j) {
                const int nt = ntb + j;
#pragma unroll
                for (int k2 = 0; k2 < 2; ++k2) {
                    const int ktg = hc * 2 + k2;
                    v16h b = *((const v16h*)w2p +
                               ((size_t)(e * 32 + ktg) * 32 + nt) * 32 + lane);
                    acc[j] = __builtin_amdgcn_wmma_f32_16x16x32_f16(
                        false, A2[k2], false, b, (short)0, acc[j], false, false);
                }
            }
        }
        __syncthreads();
    }

    // ---- Epilogue: + sum_e weights[row,e] * b2[e,col], store f32 ----
#pragma unroll
    for (int j = 0; j < 8; ++j) {
        const int nt = ntb + j;
        const int col = nt * 16 + nn;
        float b2v[E_NUM];
#pragma unroll
        for (int e = 0; e < E_NUM; ++e) b2v[e] = b2[e * D_DIM + col];
#pragma unroll
        for (int jj = 0; jj < 8; ++jj) {
            const int row = m0 + rt2 * 16 + jj + hi * 8;
            const float* wr = weights + (size_t)row * E_NUM;
            float bias = 0.f;
#pragma unroll
            for (int e = 0; e < E_NUM; ++e) bias += wr[e] * b2v[e];
            out[(size_t)row * D_DIM + col] = acc[j][jj] + bias;
        }
    }
}

// ---------------------------------------------------------------------------
extern "C" void kernel_launch(void* const* d_in, const int* in_sizes, int n_in,
                              void* d_out, int out_size, void* d_ws, size_t ws_size,
                              hipStream_t stream) {
    const float* x       = (const float*)d_in[0]; // [B,L,D]
    const float* weights = (const float*)d_in[1]; // [B,L,E]
    const float* w1      = (const float*)d_in[2]; // [E,D,H]
    const float* b1      = (const float*)d_in[3]; // [E,H]
    const float* w2      = (const float*)d_in[4]; // [E,H,D]
    const float* b2      = (const float*)d_in[5]; // [E,D]
    float* out = (float*)d_out;

    _Float16* xp  = (_Float16*)d_ws;                        // 8 MiB
    _Float16* w1p = xp  + (size_t)M_TOT * D_DIM;            // 8 MiB
    _Float16* w2p = w1p + (size_t)E_NUM * D_DIM * H_DIM;    // 8 MiB

    pack_x_kernel <<<1024, 256, 0, stream>>>(x,  xp);   // 8192 tiles
    pack_w1_kernel<<<1024, 256, 0, stream>>>(w1, w1p);  // 8192 tiles
    pack_w2_kernel<<<1024, 256, 0, stream>>>(w2, w2p);  // 8192 tiles
    moe_fused_kernel<<<M_TOT / 32, 256, 0, stream>>>(xp, w1p, w2p,
                                                     weights, b1, b2, out);
}
